// WindowAttention_89876485636822
// MI455X (gfx1250) — matrix operand order repaired and checked by reading, not compiled, not run on hardware
//
#include <hip/hip_runtime.h>

typedef __attribute__((ext_vector_type(16))) __bf16 v16bf;
typedef __attribute__((ext_vector_type(8)))  float  v8f;
typedef __attribute__((ext_vector_type(4)))  unsigned int v4u;
typedef __attribute__((ext_vector_type(4)))  int    v4i;
typedef __attribute__((ext_vector_type(8)))  int    v8i;

#define BB   2
#define CC   256
#define HH   256
#define WW_  256
#define HW   65536      // H*W
#define NH   8
#define DK   32
#define WSZ  8
#define SS   4
#define LL   64         // tokens per window
#define NW   1024       // windows per batch image
#define BW   2048       // B*NW

// ===========================================================================
// WMMA bf16 16x16x32 register layouts (CDNA5 ISA §7.12.2):
//  A (16Mx32K): lane = (m&15) + 16*((k>>3)&1); elem idx = (k&7) + 8*(k>>4)
//  B (32Kx16N), k order this code uses for A and B: lane = (n&15) + 16*(k>>4);     elem idx = k&15
//  C/D: vgpr r -> M = r + 8*(lane>=16), N = lane&15
// LDS tiles are stored "frag-ready": [tile][lane][16 bf16] so each fragment
// is one aligned 32B contiguous load (2x ds_load_b128), no scalar gathers.
// ===========================================================================

// ---------------------------------------------------------------------------
// Kernel 1: weight fp32 -> bf16. Wc = [Wq;Wkv] (768x256), Wp (256x256).
// ---------------------------------------------------------------------------
__global__ __launch_bounds__(256) void k_cvt(const float* __restrict__ Wq,
                                             const float* __restrict__ Wkv,
                                             const float* __restrict__ Wproj,
                                             __bf16* __restrict__ Wc,
                                             __bf16* __restrict__ Wp) {
  int i = blockIdx.x * 256 + threadIdx.x;
  if (i < 768 * 256) {
    float f = (i < 65536) ? Wq[i] : Wkv[i - 65536];
    Wc[i] = (__bf16)f;
  }
  if (i < 65536) Wp[i] = (__bf16)Wproj[i];
}

// ---------------------------------------------------------------------------
// Kernel 2: QKV GEMM [768x256] x [256x131072] bf16 WMMA.
// Epilogue: +bias, q *= 1/sqrt(32); scatter into rolled + window-partitioned
// FRAG-READY per-(window,head) layouts (2048 bf16 per head per tensor):
//   q: A-frag [tTile=4][lane][16]   (m=t, k=c)
//   k: B-frag [sTile=4][lane][16]   (k=c, n=s)
//   v: A-frag [ct=2][ks=2][lane][16](m=c&15, k=s&31)
// ---------------------------------------------------------------------------
__global__ __launch_bounds__(256) void k_qkv(const __bf16* __restrict__ Wc,
                                             const float* __restrict__ x,
                                             const float* __restrict__ bq,
                                             const float* __restrict__ bkv,
                                             __bf16* __restrict__ qg,
                                             __bf16* __restrict__ kg,
                                             __bf16* __restrict__ vg) {
  __shared__ v16bf lAf[256];   // 8 mTiles x 32 lanes
  __shared__ v16bf lBf[256];   // 8 nTiles x 32 lanes
  const int nT  = blockIdx.x;          // 0..1023
  const int mT  = blockIdx.y;          // 0..5
  const int tid = threadIdx.x;
  const int wave = tid >> 5;
  const int lane = tid & 31;
  const int wm = wave >> 2, wn = wave & 3;
  const int n0 = nT * 128;
  const int b  = n0 >> 16;             // tile never crosses batch (HW%128==0)
  const int p0 = n0 & 65535;

  v8f acc[4][2];
#pragma unroll
  for (int i = 0; i < 4; ++i)
#pragma unroll
    for (int j = 0; j < 2; ++j) acc[i][j] = (v8f){};

  for (int k0 = 0; k0 < 256; k0 += 32) {
    // ---- A tile: Wc rows [mT*128,+128), k in [k0,+32), frag-ready ----
    {
      int m  = tid >> 1;
      int kk = tid & 1;                      // k sub-block (0 or 16)
      const __bf16* src = Wc + (size_t)(mT * 128 + m) * 256 + k0 + kk * 16;
      uint4 u0 = *(const uint4*)(src);       // k+0..7
      uint4 u1 = *(const uint4*)(src + 8);   // k+8..15
      int baseIdx = ((m >> 4) << 5) + (m & 15);
      ((uint4*)&lAf[baseIdx + 16 * kk])[0] = u0; // lane half kk, elems 0..7
      ((uint4*)&lAf[baseIdx + 16 * kk])[1] = u1; // lane half kk, elems 8..15
    }
    // ---- B tile: x fp32, 16 channels per thread (coalesced), frag-ready ----
    {
      int pcol = tid & 127;
      int half = tid >> 7;                   // k range 16*half..+16
      const float* src = x + ((size_t)(b * CC + k0 + half * 16)) * HW + p0 + pcol;
      if (k0 + 32 < 256) __builtin_prefetch(src + (size_t)32 * HW, 0, 1);
      v16bf t;
#pragma unroll
      for (int i = 0; i < 16; ++i) t[i] = (__bf16)src[(size_t)i * HW];
      lBf[((pcol >> 4) << 5) + (pcol & 15) + 16 * half] = t;
    }
    __syncthreads();
    v16bf af[4], bfr[2];
#pragma unroll
    for (int i = 0; i < 4; ++i) af[i]  = lAf[((wm * 4 + i) << 5) + lane];
#pragma unroll
    for (int j = 0; j < 2; ++j) bfr[j] = lBf[((wn * 2 + j) << 5) + lane];
#pragma unroll
    for (int i = 0; i < 4; ++i)
#pragma unroll
      for (int j = 0; j < 2; ++j)
        acc[i][j] = __builtin_amdgcn_wmma_f32_16x16x32_bf16(
            false, af[i], false, bfr[j], (short)0, acc[i][j], false, false);
    __syncthreads();
  }

  const int hi = lane >> 4;
  const float scale = 0.17677669529663687f;  // 32^-0.5
#pragma unroll
  for (int i = 0; i < 4; ++i)
#pragma unroll
    for (int j = 0; j < 2; ++j)
#pragma unroll
      for (int r = 0; r < 8; ++r) {
        int o = mT * 128 + wm * 64 + i * 16 + r + hi * 8;
        int n = n0 + wn * 32 + j * 16 + (lane & 15);
        float val = acc[i][j][r] + ((o < 256) ? bq[o] : bkv[o - 256]);
        if (o < 256) val *= scale;
        int which = o >> 8;
        int oc = o & 255;
        int head = oc >> 5, c = oc & 31;
        int pp = n & 65535;
        int h = pp >> 8, w = pp & 255;
        int hr = (h - SS) & 255, wr = (w - SS) & 255;   // roll(-4,-4)
        int wi = ((hr >> 3) << 5) + (wr >> 3);
        int t  = ((hr & 7) << 3) + (wr & 7);
        size_t hb = (((size_t)(b * NW + wi) * NH + head) << 11);  // *2048
        __bf16 bv = (__bf16)val;
        if (which == 0) {
          size_t off = hb + ((size_t)(t >> 4) << 9)
                     + (((t & 15) + ((c >> 4) << 4)) << 4)
                     + (c & 15);
          qg[off] = bv;
        } else if (which == 1) {
          size_t off = hb + ((size_t)(t >> 4) << 9)
                     + (((t & 15) + ((c >> 4) << 4)) << 4)
                     + (c & 15);
          kg[off] = bv;
        } else {
          size_t off = hb + ((size_t)(((c >> 4) << 1) + (t >> 5)) << 9)
                     + (((c & 15) + (((t >> 4) & 1) << 4)) << 4)
                     + (t & 15);
          vg[off] = bv;
        }
      }
}

// ---------------------------------------------------------------------------
// TDM 1-D descriptor load: contiguous nelem (2-byte elems) -> LDS offset.
// D# per CDNA5 ISA. Guarded per-toolchain arity.
// ---------------------------------------------------------------------------
#if __has_builtin(__builtin_amdgcn_tensor_load_to_lds)
#define HAVE_TDM 1
__device__ __forceinline__ void tdm_load_1d(unsigned lds_off_bytes,
                                            const void* gptr, unsigned nelem) {
  unsigned long long ga = (unsigned long long)gptr;
  v4u g0 = { 1u,                               // count=1, user mode
             lds_off_bytes,                    // lds_addr [63:32]
             (unsigned)ga,                     // global_addr [95:64]
             (unsigned)((ga >> 32) & 0x01FFFFFFu) | (2u << 30) }; // +type=2
  v8i g1 = { (int)(1u << 16),                  // data_size=1 (2B)
             (int)((nelem & 0xFFFFu) << 16),   // tensor_dim0[15:0] @bit48
             (int)((nelem >> 16) | (1u << 16)),// tensor_dim0 hi | tensor_dim1=1
             (int)((nelem & 0xFFFFu) << 16),   // tile_dim0 @bit112
             1,                                // tile_dim1=1
             (int)nelem, 0, 0 };               // tensor_dim0_stride
  v4i gz = { 0, 0, 0, 0 };
#if __clang_major__ >= 23
  v8i gz8 = { 0, 0, 0, 0, 0, 0, 0, 0 };
  __builtin_amdgcn_tensor_load_to_lds(g0, g1, gz, gz, gz8, 0);
#else
  __builtin_amdgcn_tensor_load_to_lds(g0, g1, gz, gz, 0);
#endif
}
#else
#define HAVE_TDM 0
#endif

// ---------------------------------------------------------------------------
// Kernel 3: window attention. 1 block/window (2048), 1 wave/head.
// Dynamic LDS 96KB: q | k | v frag-ready (2048 bf16 per head each).
// P (softmaxed attn) reuses q+k region in B-frag layout [tTile][ks][lane][16].
// ---------------------------------------------------------------------------
__global__ __launch_bounds__(256) void k_attn(const __bf16* __restrict__ qg,
                                              const __bf16* __restrict__ kg,
                                              const __bf16* __restrict__ vg,
                                              const float* __restrict__ bias_table,
                                              __bf16* __restrict__ y) {
  extern __shared__ __attribute__((aligned(32))) char smem_raw[];
  __bf16* lds = (__bf16*)smem_raw;               // 3*16384 bf16
  const int bw = blockIdx.x;
  const int b  = bw >> 10, wi = bw & 1023;
  const int tid = threadIdx.x;

#if HAVE_TDM
  if (tid < 32) {  // one wave issues the three TDM descriptors
    tdm_load_1d(0u,     qg + (size_t)bw * 16384, 16384u);
    tdm_load_1d(32768u, kg + (size_t)bw * 16384, 16384u);
    tdm_load_1d(65536u, vg + (size_t)bw * 16384, 16384u);
    __builtin_amdgcn_s_wait_tensorcnt(0);
  }
#else
  {
    const uint4* sq = (const uint4*)(qg + (size_t)bw * 16384);
    const uint4* sk = (const uint4*)(kg + (size_t)bw * 16384);
    const uint4* sv = (const uint4*)(vg + (size_t)bw * 16384);
    uint4* d = (uint4*)lds;
#pragma unroll 2
    for (int i = tid; i < 2048; i += 256) {
      d[i]        = sq[i];
      d[2048 + i] = sk[i];
      d[4096 + i] = sv[i];
    }
  }
#endif
  __syncthreads();

  const int head = tid >> 5;
  const int lane = tid & 31;
  const int hi   = lane >> 4;
  const v16bf* fr = (const v16bf*)lds;           // frag view of LDS

  // ---- S = q^T k : one K=32 step, 16 WMMAs ----
  v16bf aq[4], bk[4];
#pragma unroll
  for (int mt = 0; mt < 4; ++mt) aq[mt] = fr[head * 128 + (mt << 5) + lane];
#pragma unroll
  for (int nt = 0; nt < 4; ++nt) bk[nt] = fr[1024 + head * 128 + (nt << 5) + lane];
  v8f att[4][4];
#pragma unroll
  for (int mt = 0; mt < 4; ++mt)
#pragma unroll
    for (int nt = 0; nt < 4; ++nt)
      att[mt][nt] = __builtin_amdgcn_wmma_f32_16x16x32_bf16(
          false, aq[mt], false, bk[nt], (short)0, (v8f){}, false, false);

  // ---- bias + shift-mask + row softmax (rows live in 16-lane halves) ----
  const int wh = wi >> 5, ww = wi & 31;
#pragma unroll
  for (int mt = 0; mt < 4; ++mt) {
#pragma unroll
    for (int r = 0; r < 8; ++r) {
      int t  = mt * 16 + r + hi * 8;
      int th = t >> 3, tw = t & 7;
      int gh = wh * 8 + th, gw = ww * 8 + tw;
      int rhT = (gh < HH - WSZ) ? 0 : ((gh < HH - SS) ? 1 : 2);
      int rwT = (gw < WW_ - WSZ) ? 0 : ((gw < WW_ - SS) ? 1 : 2);
      int idT = rhT * 3 + rwT;
      float vals[4];
#pragma unroll
      for (int nt = 0; nt < 4; ++nt) {
        int s  = nt * 16 + (lane & 15);
        int sh = s >> 3, sw = s & 7;
        int ridx = (th - sh + 7) * 15 + (tw - sw + 7);
        float bb = bias_table[ridx * NH + head];
        int sgh = wh * 8 + sh, sgw = ww * 8 + sw;
        int rhS = (sgh < HH - WSZ) ? 0 : ((sgh < HH - SS) ? 1 : 2);
        int rwS = (sgw < WW_ - WSZ) ? 0 : ((sgw < WW_ - SS) ? 1 : 2);
        float mv = (idT != rhS * 3 + rwS) ? -100.0f : 0.0f;
        vals[nt] = att[mt][nt][r] + bb + mv;
      }
      float mx = fmaxf(fmaxf(vals[0], vals[1]), fmaxf(vals[2], vals[3]));
#pragma unroll
      for (int m2 = 1; m2 < 16; m2 <<= 1) mx = fmaxf(mx, __shfl_xor(mx, m2, 32));
      float sum = 0.f;
#pragma unroll
      for (int nt = 0; nt < 4; ++nt) { vals[nt] = __expf(vals[nt] - mx); sum += vals[nt]; }
#pragma unroll
      for (int m2 = 1; m2 < 16; m2 <<= 1) sum += __shfl_xor(sum, m2, 32);
      float inv = 1.0f / sum;
#pragma unroll
      for (int nt = 0; nt < 4; ++nt) att[mt][nt][r] = vals[nt] * inv;
    }
  }

  // ---- write P to LDS in B-frag layout (reuses q+k region) ----
  __syncthreads();                               // all waves done reading q,k
  __bf16* latt = lds + head * 4096;              // [tTile][ks][lane][16]
#pragma unroll
  for (int mt = 0; mt < 4; ++mt)
#pragma unroll
    for (int nt = 0; nt < 4; ++nt)
#pragma unroll
      for (int r = 0; r < 8; ++r) {
        int t = mt * 16 + r + hi * 8;
        int s = nt * 16 + (lane & 15);
        int off = ((((t >> 4) << 1) + (s >> 5)) << 9)
                + (((t & 15) + (((s >> 4) & 1) << 4)) << 4)
                + (s & 15);
        latt[off] = (__bf16)att[mt][nt][r];
      }
  __syncthreads();

  // ---- O = v P^T : K=64 -> 2 steps, 16 WMMAs ----
  v8f outacc[2][4];
#pragma unroll
  for (int ct = 0; ct < 2; ++ct)
#pragma unroll
    for (int nt = 0; nt < 4; ++nt) outacc[ct][nt] = (v8f){};
#pragma unroll
  for (int ks = 0; ks < 2; ++ks) {
    v16bf av[2], ba[4];
#pragma unroll
    for (int ct = 0; ct < 2; ++ct)
      av[ct] = fr[2048 + head * 128 + (((ct << 1) + ks) << 5) + lane];
#pragma unroll
    for (int nt = 0; nt < 4; ++nt)
      ba[nt] = fr[head * 256 + (((nt << 1) + ks) << 5) + lane];
#pragma unroll
    for (int ct = 0; ct < 2; ++ct)
#pragma unroll
      for (int nt = 0; nt < 4; ++nt)
        outacc[ct][nt] = __builtin_amdgcn_wmma_f32_16x16x32_bf16(
            false, av[ct], false, ba[nt], (short)0, outacc[ct][nt], false, false);
  }

  // ---- un-partition + roll(+4,+4), store bf16 y[b][ch][h][w] ----
#pragma unroll
  for (int ct = 0; ct < 2; ++ct)
#pragma unroll
    for (int nt = 0; nt < 4; ++nt)
#pragma unroll
      for (int r = 0; r < 8; ++r) {
        int c  = ct * 16 + r + hi * 8;
        int t  = nt * 16 + (lane & 15);
        int ch = head * DK + c;
        int th = t >> 3, tw = t & 7;
        int h  = ((wh * 8 + th) + SS) & 255;
        int w  = ((ww * 8 + tw) + SS) & 255;
        y[((size_t)(b * CC + ch) << 16) + (h << 8) + w] = (__bf16)outacc[ct][nt][r];
      }
}

// ---------------------------------------------------------------------------
// Kernel 4: output projection [256x256] x [256x131072] bf16 -> fp32 (+bias).
// ---------------------------------------------------------------------------
__global__ __launch_bounds__(256) void k_proj(const __bf16* __restrict__ Wp,
                                              const __bf16* __restrict__ y,
                                              const float* __restrict__ bproj,
                                              float* __restrict__ out) {
  __shared__ v16bf lAf[256];
  __shared__ v16bf lBf[256];
  const int nT  = blockIdx.x;
  const int mT  = blockIdx.y;          // 0..1
  const int tid = threadIdx.x;
  const int wave = tid >> 5;
  const int lane = tid & 31;
  const int wm = wave >> 2, wn = wave & 3;
  const int n0 = nT * 128;
  const int b  = n0 >> 16;
  const int p0 = n0 & 65535;

  v8f acc[4][2];
#pragma unroll
  for (int i = 0; i < 4; ++i)
#pragma unroll
    for (int j = 0; j < 2; ++j) acc[i][j] = (v8f){};

  for (int k0 = 0; k0 < 256; k0 += 32) {
    {
      int m  = tid >> 1;
      int kk = tid & 1;
      const __bf16* src = Wp + (size_t)(mT * 128 + m) * 256 + k0 + kk * 16;
      uint4 u0 = *(const uint4*)(src);
      uint4 u1 = *(const uint4*)(src + 8);
      int baseIdx = ((m >> 4) << 5) + (m & 15);
      ((uint4*)&lAf[baseIdx + 16 * kk])[0] = u0;
      ((uint4*)&lAf[baseIdx + 16 * kk])[1] = u1;
    }
    {
      int pcol = tid & 127;
      int half = tid >> 7;
      const __bf16* src = y + ((size_t)(b * CC + k0 + half * 16)) * HW + p0 + pcol;
      if (k0 + 32 < 256) __builtin_prefetch(src + (size_t)32 * HW, 0, 1);
      v16bf t;
#pragma unroll
      for (int i = 0; i < 16; ++i) t[i] = src[(size_t)i * HW];
      lBf[((pcol >> 4) << 5) + (pcol & 15) + 16 * half] = t;
    }
    __syncthreads();
    v16bf af[4], bfr[2];
#pragma unroll
    for (int i = 0; i < 4; ++i) af[i]  = lAf[((wm * 4 + i) << 5) + lane];
#pragma unroll
    for (int j = 0; j < 2; ++j) bfr[j] = lBf[((wn * 2 + j) << 5) + lane];
#pragma unroll
    for (int i = 0; i < 4; ++i)
#pragma unroll
      for (int j = 0; j < 2; ++j)
        acc[i][j] = __builtin_amdgcn_wmma_f32_16x16x32_bf16(
            false, af[i], false, bfr[j], (short)0, acc[i][j], false, false);
    __syncthreads();
  }

  const int hi = lane >> 4;
#pragma unroll
  for (int i = 0; i < 4; ++i)
#pragma unroll
    for (int j = 0; j < 2; ++j)
#pragma unroll
      for (int r = 0; r < 8; ++r) {
        int o = mT * 128 + wm * 64 + i * 16 + r + hi * 8;
        int n = n0 + wn * 32 + j * 16 + (lane & 15);
        int pp = n & 65535;
        out[((size_t)(b * CC + o) << 16) + pp] = acc[i][j][r] + bproj[o];
      }
}

// ---------------------------------------------------------------------------
extern "C" void kernel_launch(void* const* d_in, const int* in_sizes, int n_in,
                              void* d_out, int out_size, void* d_ws, size_t ws_size,
                              hipStream_t stream) {
  const float* x          = (const float*)d_in[0];
  const float* Wq         = (const float*)d_in[1];
  const float* bq         = (const float*)d_in[2];
  const float* Wkv        = (const float*)d_in[3];
  const float* bkv        = (const float*)d_in[4];
  const float* Wproj      = (const float*)d_in[5];
  const float* bproj      = (const float*)d_in[6];
  const float* bias_table = (const float*)d_in[7];

  char* ws = (char*)d_ws;
  size_t off = 0;
  auto take = [&](size_t bytes) -> char* {
    char* p = ws + off;
    off = (off + bytes + 255) & ~(size_t)255;
    return p;
  };
  __bf16* Wc = (__bf16*)take((size_t)768 * 256 * 2);
  __bf16* Wp = (__bf16*)take((size_t)256 * 256 * 2);
  __bf16* qb = (__bf16*)take((size_t)BW * NH * DK * LL * 2);
  __bf16* kb = (__bf16*)take((size_t)BW * NH * DK * LL * 2);
  __bf16* vb = (__bf16*)take((size_t)BW * NH * DK * LL * 2);
  __bf16* yb = (__bf16*)take((size_t)BB * CC * HW * 2);

  k_cvt <<<768, 256, 0, stream>>>(Wq, Wkv, Wproj, Wc, Wp);
  k_qkv <<<dim3(1024, 6), 256, 0, stream>>>(Wc, x, bq, bkv, qb, kb, vb);
  k_attn<<<BW, 256, 3 * 16384 * 2, stream>>>(qb, kb, vb, bias_table, yb);
  k_proj<<<dim3(1024, 2), 256, 0, stream>>>(Wp, yb, bproj, (float*)d_out);
}
